// ASPPDeformableC3D_53300544143738
// MI455X (gfx1250) — matrix operand order repaired and checked by reading, not compiled, not run on hardware
//
#include <hip/hip_runtime.h>

typedef _Float16 half_t;
typedef __attribute__((ext_vector_type(16))) _Float16 v16h;
typedef __attribute__((ext_vector_type(8)))  _Float16 v8h;
typedef __attribute__((ext_vector_type(8)))  float    v8f;

#define NSPAT 25088   // 8*56*56
#define DD 8
#define HH 56
#define WW 56

// ---------------------------------------------------------------------------
// Prep: transpose x [C=128][N] f32 -> xT [N][128] f16
// ---------------------------------------------------------------------------
__global__ void k_transpose_x(const float* __restrict__ x, half_t* __restrict__ xT) {
    int n = blockIdx.x * 256 + threadIdx.x;          // 98*256 == 25088 exactly
    #pragma unroll 4
    for (int c = 0; c < 128; ++c)
        xT[(size_t)n * 128 + c] = (half_t)x[(size_t)c * NSPAT + n];
}

// per-channel spatial mean of x (for the global-pool branch)
__global__ void k_channel_mean(const float* __restrict__ x, float* __restrict__ mean) {
    __shared__ float sm[256];
    int c = blockIdx.x;
    float s = 0.f;
    for (int n = threadIdx.x; n < NSPAT; n += 256) s += x[(size_t)c * NSPAT + n];
    sm[threadIdx.x] = s;
    __syncthreads();
    for (int off = 128; off > 0; off >>= 1) {
        if (threadIdx.x < off) sm[threadIdx.x] += sm[threadIdx.x + off];
        __syncthreads();
    }
    if (threadIdx.x == 0) mean[c] = sm[0] / (float)NSPAT;
}

// g = relu(Wg@mean + bg);  constv[m] = bp[m] + sum_j wp[m][1024+j]*g[j]
__global__ void k_pool_const(const float* __restrict__ mean,
                             const float* __restrict__ wg, const float* __restrict__ bg,
                             const float* __restrict__ wp, const float* __restrict__ bp,
                             float* __restrict__ constv) {
    __shared__ float g[256];
    int m = threadIdx.x;                              // 256 threads
    float s = bg[m];
    for (int c = 0; c < 128; ++c) s += wg[m * 128 + c] * mean[c];
    g[m] = fmaxf(s, 0.f);
    __syncthreads();
    float t = bp[m];
    for (int j = 0; j < 256; ++j) t += wp[(size_t)m * 1280 + 1024 + j] * g[j];
    constv[m] = t;
}

__global__ void k_pad_bias(const float* __restrict__ b, float* __restrict__ bpad,
                           int nreal, int npad) {
    int i = blockIdx.x * 64 + threadIdx.x;
    if (i < npad) bpad[i] = (i < nreal) ? b[i] : 0.f;
}

// ---------------------------------------------------------------------------
// Pack f32 weights into WMMA A-fragment order (f16).
// Fragment frag = ((tap*MT + mt)*KT + kt), 512 halves: lane*16 + e.
// A layout: lanes<16 hold K={0..15}, lanes>=16 K={16..31} (matches contiguous B).
// src element (m,c,tap) at  m*strideM + c*strideC + tap*strideT.
// ---------------------------------------------------------------------------
__global__ __launch_bounds__(32)
void k_pack_a(const float* __restrict__ src, half_t* __restrict__ dst,
              int Mreal, int MT, int KT,
              long strideM, long strideC, long strideT) {
    int frag = blockIdx.x;
    int kt = frag % KT; int t2 = frag / KT;
    int mt = t2 % MT;   int tap = t2 / MT;
    int lane = threadIdx.x;
    int hi = lane >> 4;
    int m = mt * 16 + (lane & 15);
    half_t* out = dst + (size_t)frag * 512 + lane * 16;
    #pragma unroll
    for (int e = 0; e < 16; ++e) {
        int K = e + (hi ? 16 : 0);
        int c = kt * 32 + K;
        float v = (m < Mreal)
                    ? src[(size_t)m * strideM + (size_t)c * strideC + (size_t)tap * strideT]
                    : 0.f;
        out[e] = (half_t)v;
    }
}

// ---------------------------------------------------------------------------
// Implicit-GEMM conv core. One wave per block: 4x16-col N tiles x 2 M tiles,
// so every A fragment load is reused by 4 B fragments (4x less L2 A-traffic).
// src: [N][CS] f16.  apack: A fragments, taps*MT*KT.  Out: relu-f16 [N][MTOT]
// or raw f32 [N][MTOT] (offsets path).
// ---------------------------------------------------------------------------
template <int CS, int TAPS, int MTOT, bool F32OUT>
__global__ __launch_bounds__(32)
void k_conv_gemm(const half_t* __restrict__ src,
                 const half_t* __restrict__ apack,
                 const float* __restrict__ bias,
                 half_t* __restrict__ outh,
                 float* __restrict__ outf,
                 int dil) {
    constexpr int KT = CS / 32;
    constexpr int MT = MTOT / 16;
    const int lane = threadIdx.x;
    const int hi = lane >> 4;
    const int col = lane & 15;
    const int mt0 = blockIdx.y * 2;

    int n[4], d[4], h[4], w[4];
    #pragma unroll
    for (int j = 0; j < 4; ++j) {
        n[j] = blockIdx.x * 64 + j * 16 + col;
        d[j] = n[j] / (HH * WW);
        int rem = n[j] - d[j] * (HH * WW);
        h[j] = rem / WW;
        w[j] = rem - h[j] * WW;
    }

    v8f acc[4][2];
    #pragma unroll
    for (int j = 0; j < 4; ++j) { acc[j][0] = (v8f){}; acc[j][1] = (v8f){}; }

    #pragma unroll 1
    for (int t = 0; t < TAPS; ++t) {
        int sd = 0, sh = 0, sw = 0;
        if (TAPS == 27) {
            int kd = t / 9, kh = (t / 3) % 3, kw = t % 3;
            sd = (kd - 1) * dil; sh = (kh - 1) * dil; sw = (kw - 1) * dil;
        }
        int lin[4]; bool valid[4];
        #pragma unroll
        for (int j = 0; j < 4; ++j) {
            int dd = d[j] + sd, hh2 = h[j] + sh, ww2 = w[j] + sw;
            valid[j] = (dd >= 0) & (dd < DD) & (hh2 >= 0) & (hh2 < HH) &
                       (ww2 >= 0) & (ww2 < WW);
            lin[j] = (dd * HH + hh2) * WW + ww2;
        }
        #pragma unroll
        for (int kt = 0; kt < KT; ++kt) {
            v16h A0 = *(const v16h*)(apack +
                (((size_t)t * MT + (mt0 + 0)) * KT + kt) * 512 + lane * 16);
            v16h A1 = *(const v16h*)(apack +
                (((size_t)t * MT + (mt0 + 1)) * KT + kt) * 512 + lane * 16);
            #pragma unroll
            for (int j = 0; j < 4; ++j) {
                v16h b = {};
                if (valid[j])
                    b = *(const v16h*)(src + (size_t)lin[j] * CS + kt * 32 + hi * 16);
                acc[j][0] = __builtin_amdgcn_wmma_f32_16x16x32_f16(
                    false, A0, false, b, (short)0, acc[j][0], false, false);
                acc[j][1] = __builtin_amdgcn_wmma_f32_16x16x32_f16(
                    false, A1, false, b, (short)0, acc[j][1], false, false);
            }
        }
    }

    #pragma unroll
    for (int j = 0; j < 4; ++j)
    #pragma unroll
    for (int which = 0; which < 2; ++which) {
        v8f acc1 = acc[j][which];
        int mbase = (mt0 + which) * 16 + hi * 8;
        if (F32OUT) {
            float4 lo, hiq;
            lo.x = acc1[0] + bias[mbase + 0]; lo.y = acc1[1] + bias[mbase + 1];
            lo.z = acc1[2] + bias[mbase + 2]; lo.w = acc1[3] + bias[mbase + 3];
            hiq.x = acc1[4] + bias[mbase + 4]; hiq.y = acc1[5] + bias[mbase + 5];
            hiq.z = acc1[6] + bias[mbase + 6]; hiq.w = acc1[7] + bias[mbase + 7];
            *(float4*)(outf + (size_t)n[j] * MTOT + mbase) = lo;
            *(float4*)(outf + (size_t)n[j] * MTOT + mbase + 4) = hiq;
        } else {
            v8h hres;
            #pragma unroll
            for (int r = 0; r < 8; ++r) {
                float v = acc1[r] + bias[mbase + r];
                hres[r] = (half_t)fmaxf(v, 0.f);
            }
            *(v8h*)(outh + (size_t)n[j] * MTOT + mbase) = hres;
        }
    }
}

// ---------------------------------------------------------------------------
// Projection: P = relu( sum_b Wp_b @ Yb + constv )  (pool branch + bias folded
// into constv).  4 N-tiles x 2 M-tiles per wave, 4 "taps" = 4 sources.
// Source loop fully unrolled so each Yb pointer is used directly -> compiler
// can prove global address space (global_load_b128, no flat/DScnt path).
// ---------------------------------------------------------------------------
__global__ __launch_bounds__(32)
void k_proj_gemm(const half_t* __restrict__ Y0, const half_t* __restrict__ Y1,
                 const half_t* __restrict__ Y2, const half_t* __restrict__ Y3,
                 const half_t* __restrict__ wpp,
                 const float* __restrict__ constv,
                 half_t* __restrict__ P) {
    const int lane = threadIdx.x;
    const int hi = lane >> 4;
    const int col = lane & 15;
    const int mt0 = blockIdx.y * 2;

    int n[4];
    #pragma unroll
    for (int j = 0; j < 4; ++j) n[j] = blockIdx.x * 64 + j * 16 + col;

    v8f acc[4][2];
    #pragma unroll
    for (int j = 0; j < 4; ++j) { acc[j][0] = (v8f){}; acc[j][1] = (v8f){}; }

    #pragma unroll
    for (int b = 0; b < 4; ++b) {
        const half_t* sb = (b == 0) ? Y0 : (b == 1) ? Y1 : (b == 2) ? Y2 : Y3;
        #pragma unroll
        for (int kt = 0; kt < 8; ++kt) {
            v16h A0 = *(const v16h*)(wpp +
                (((size_t)b * 16 + mt0 + 0) * 8 + kt) * 512 + lane * 16);
            v16h A1 = *(const v16h*)(wpp +
                (((size_t)b * 16 + mt0 + 1) * 8 + kt) * 512 + lane * 16);
            #pragma unroll
            for (int j = 0; j < 4; ++j) {
                v16h bf = *(const v16h*)(sb + (size_t)n[j] * 256 + kt * 32 + hi * 16);
                acc[j][0] = __builtin_amdgcn_wmma_f32_16x16x32_f16(
                    false, A0, false, bf, (short)0, acc[j][0], false, false);
                acc[j][1] = __builtin_amdgcn_wmma_f32_16x16x32_f16(
                    false, A1, false, bf, (short)0, acc[j][1], false, false);
            }
        }
    }
    #pragma unroll
    for (int j = 0; j < 4; ++j)
    #pragma unroll
    for (int which = 0; which < 2; ++which) {
        v8f acc1 = acc[j][which];
        int mbase = (mt0 + which) * 16 + hi * 8;
        v8h hres;
        #pragma unroll
        for (int r = 0; r < 8; ++r)
            hres[r] = (half_t)fmaxf(acc1[r] + constv[mbase + r], 0.f);
        *(v8h*)(P + (size_t)n[j] * 256 + mbase) = hres;
    }
}

// ---------------------------------------------------------------------------
// Fused deformable conv: per (tap k, N-tile pair) build trilinearly-sampled
// B fragments from xT with packed f16 FMAs (v_pk_fma_f16), then 8 M-tiles of
// WMMA against packed wc, each A fragment reused by both N-tiles.
// Output NCDHW f32 + bc.
// ---------------------------------------------------------------------------
__global__ __launch_bounds__(32)
void k_deform_gemm(const half_t* __restrict__ xT,   // [N][128]
                   const float* __restrict__ off,   // [N][96]  (k*3+comp, k<27)
                   const half_t* __restrict__ wcp,  // taps=27, MT=8, KT=4
                   const float* __restrict__ bc,
                   float* __restrict__ out) {       // [128][N]
    const int lane = threadIdx.x;
    const int hi = lane >> 4;
    const int col = lane & 15;

    int n[2], d[2], h[2], w[2];
    #pragma unroll
    for (int j = 0; j < 2; ++j) {
        n[j] = blockIdx.x * 32 + j * 16 + col;
        d[j] = n[j] / (HH * WW);
        int rem = n[j] - d[j] * (HH * WW);
        h[j] = rem / WW;
        w[j] = rem - h[j] * WW;
    }

    v8f acc[2][8];
    #pragma unroll
    for (int j = 0; j < 2; ++j)
    #pragma unroll
    for (int mt = 0; mt < 8; ++mt) acc[j][mt] = (v8f){};

    #pragma unroll 1
    for (int k = 0; k < 27; ++k) {
        int kd = k / 9 - 1, kh = (k / 3) % 3 - 1, kw = k % 3 - 1;

        half_t whf[2][8]; int lin[2][8];
        #pragma unroll
        for (int j = 0; j < 2; ++j) {
            const float* o = off + (size_t)n[j] * 96 + k * 3;
            float pd = (float)(d[j] + kd) + o[0];
            float ph = (float)(h[j] + kh) + o[1];
            float pw = (float)(w[j] + kw) + o[2];
            float d0f = floorf(pd), fd = pd - d0f; int d0 = (int)d0f;
            float h0f = floorf(ph), fh = ph - h0f; int h0 = (int)h0f;
            float w0f = floorf(pw), fw = pw - w0f; int w0 = (int)w0f;
            #pragma unroll
            for (int cd = 0; cd < 2; ++cd)
            #pragma unroll
            for (int ch = 0; ch < 2; ++ch)
            #pragma unroll
            for (int cw = 0; cw < 2; ++cw) {
                int idx = cd * 4 + ch * 2 + cw;
                int di = d0 + cd, hi2 = h0 + ch, wi = w0 + cw;
                bool val = (di >= 0) & (di < DD) & (hi2 >= 0) & (hi2 < HH) &
                           (wi >= 0) & (wi < WW);
                float wt = (cd ? fd : 1.f - fd) * (ch ? fh : 1.f - fh) *
                           (cw ? fw : 1.f - fw);
                whf[j][idx] = (half_t)(val ? wt : 0.f);
                int dc = min(max(di, 0), DD - 1);
                int hc = min(max(hi2, 0), HH - 1);
                int wc_ = min(max(wi, 0), WW - 1);
                lin[j][idx] = (dc * HH + hc) * WW + wc_;
            }
        }

        #pragma unroll
        for (int kt = 0; kt < 4; ++kt) {
            v16h bfr[2];
            #pragma unroll
            for (int j = 0; j < 2; ++j) {
                v16h bacc = {};
                #pragma unroll
                for (int c8 = 0; c8 < 8; ++c8) {
                    if ((float)whf[j][c8] != 0.f) {
                        v16h xv = *(const v16h*)(xT + (size_t)lin[j][c8] * 128 +
                                                 kt * 32 + hi * 16);
                        v16h ws;
                        #pragma unroll
                        for (int e = 0; e < 16; ++e) ws[e] = whf[j][c8];
                        bacc += ws * xv;                 // v_pk_fma_f16
                    }
                }
                bfr[j] = bacc;
            }
            #pragma unroll
            for (int mt = 0; mt < 8; ++mt) {
                v16h A = *(const v16h*)(wcp +
                    (((size_t)k * 8 + mt) * 4 + kt) * 512 + lane * 16);
                acc[0][mt] = __builtin_amdgcn_wmma_f32_16x16x32_f16(
                    false, A, false, bfr[0], (short)0, acc[0][mt], false, false);
                acc[1][mt] = __builtin_amdgcn_wmma_f32_16x16x32_f16(
                    false, A, false, bfr[1], (short)0, acc[1][mt], false, false);
            }
        }
    }

    #pragma unroll
    for (int j = 0; j < 2; ++j)
    #pragma unroll
    for (int mt = 0; mt < 8; ++mt) {
        int mbase = mt * 16 + hi * 8;
        #pragma unroll
        for (int r = 0; r < 8; ++r)
            out[(size_t)(mbase + r) * NSPAT + n[j]] = acc[j][mt][r] + bc[mbase + r];
    }
}

// ---------------------------------------------------------------------------
extern "C" void kernel_launch(void* const* d_in, const int* in_sizes, int n_in,
                              void* d_out, int out_size, void* d_ws, size_t ws_size,
                              hipStream_t stream) {
    (void)in_sizes; (void)n_in; (void)out_size; (void)ws_size;
    const float* x  = (const float*)d_in[0];
    const float* w0 = (const float*)d_in[1];  const float* b0 = (const float*)d_in[2];
    const float* w1 = (const float*)d_in[3];  const float* b1 = (const float*)d_in[4];
    const float* w2 = (const float*)d_in[5];  const float* b2 = (const float*)d_in[6];
    const float* w3 = (const float*)d_in[7];  const float* b3 = (const float*)d_in[8];
    const float* wg = (const float*)d_in[9];  const float* bg = (const float*)d_in[10];
    const float* wp = (const float*)d_in[11]; const float* bp = (const float*)d_in[12];
    const float* wd = (const float*)d_in[13]; const float* bd = (const float*)d_in[14];
    const float* wc = (const float*)d_in[15]; const float* bc = (const float*)d_in[16];
    float* out = (float*)d_out;

    char* ws = (char*)d_ws;
    size_t o = 0;
    auto alloc = [&](size_t bytes) -> void* {
        void* p = ws + o;
        o = (o + bytes + 255) & ~(size_t)255;
        return p;
    };

    half_t* xT  = (half_t*)alloc((size_t)NSPAT * 128 * 2);
    half_t* Y0  = (half_t*)alloc((size_t)NSPAT * 256 * 2);
    half_t* Y1  = (half_t*)alloc((size_t)NSPAT * 256 * 2);
    half_t* Y2  = (half_t*)alloc((size_t)NSPAT * 256 * 2);
    half_t* Y3  = (half_t*)alloc((size_t)NSPAT * 256 * 2);
    half_t* P   = (half_t*)alloc((size_t)NSPAT * 256 * 2);
    float*  OFF = (float*) alloc((size_t)NSPAT * 96 * 4);
    half_t* w0p = (half_t*)alloc((size_t)64   * 1024);   //  1*16*4 frags
    half_t* w1p = (half_t*)alloc((size_t)1728 * 1024);   // 27*16*4
    half_t* w2p = (half_t*)alloc((size_t)1728 * 1024);
    half_t* w3p = (half_t*)alloc((size_t)1728 * 1024);
    half_t* wpp = (half_t*)alloc((size_t)512  * 1024);   //  4*16*8
    half_t* wdp = (half_t*)alloc((size_t)1296 * 1024);   // 27*6*8
    half_t* wcp = (half_t*)alloc((size_t)864  * 1024);   // 27*8*4
    float* meanb  = (float*)alloc(128 * 4);
    float* constv = (float*)alloc(256 * 4);
    float* bdpad  = (float*)alloc(96 * 4);

    // ---- prep ----
    k_transpose_x<<<NSPAT / 256, 256, 0, stream>>>(x, xT);
    k_channel_mean<<<128, 256, 0, stream>>>(x, meanb);
    k_pool_const<<<1, 256, 0, stream>>>(meanb, wg, bg, wp, bp, constv);
    k_pad_bias<<<2, 64, 0, stream>>>(bd, bdpad, 81, 96);

    // ---- weight packs (src strides: m, c, tap) ----
    k_pack_a<<<64,   32, 0, stream>>>(w0, w0p, 256, 16, 4, 128,  1, 0);
    k_pack_a<<<1728, 32, 0, stream>>>(w1, w1p, 256, 16, 4, 3456, 27, 1);
    k_pack_a<<<1728, 32, 0, stream>>>(w2, w2p, 256, 16, 4, 3456, 27, 1);
    k_pack_a<<<1728, 32, 0, stream>>>(w3, w3p, 256, 16, 4, 3456, 27, 1);
    k_pack_a<<<512,  32, 0, stream>>>(wp, wpp, 256, 16, 8, 1280, 1, 256);
    k_pack_a<<<1296, 32, 0, stream>>>(wd, wdp, 81,  6,  8, 6912, 27, 1);
    k_pack_a<<<864,  32, 0, stream>>>(wc, wcp, 128, 8,  4, 3456, 27, 1);

    // ---- ASPP branches (implicit GEMM, WMMA f16->f32) ----
    dim3 g256(NSPAT / 64, 8);                 // 392 x 8
    k_conv_gemm<128, 1,  256, false><<<g256, 32, 0, stream>>>(xT, w0p, b0, Y0, nullptr, 1);
    k_conv_gemm<128, 27, 256, false><<<g256, 32, 0, stream>>>(xT, w1p, b1, Y1, nullptr, 2);
    k_conv_gemm<128, 27, 256, false><<<g256, 32, 0, stream>>>(xT, w2p, b2, Y2, nullptr, 4);
    k_conv_gemm<128, 27, 256, false><<<g256, 32, 0, stream>>>(xT, w3p, b3, Y3, nullptr, 8);

    // ---- projection (concat folded; pool branch in constv) ----
    k_proj_gemm<<<g256, 32, 0, stream>>>(Y0, Y1, Y2, Y3, wpp, constv, P);

    // ---- offset conv (pad=1), f32 out [N][96] ----
    dim3 g96(NSPAT / 64, 3);                  // 392 x 3
    k_conv_gemm<256, 27, 96, true><<<g96, 32, 0, stream>>>(P, wdp, bdpad, nullptr, OFF, 1);

    // ---- fused deformable conv -> d_out (NCDHW f32) ----
    k_deform_gemm<<<NSPAT / 32, 32, 0, stream>>>(xT, OFF, wcp, bc, out);
}
